// MAN_84825604096408
// MI455X (gfx1250) — matrix operand order repaired and checked by reading, not compiled, not run on hardware
//
#include <hip/hip_runtime.h>
#include <hip/hip_bf16.h>
#include <math.h>
#include <stddef.h>

// ---------------- types ----------------
typedef __attribute__((ext_vector_type(16))) __bf16 v16bf;
typedef __attribute__((ext_vector_type(8)))  __bf16 bf16x8;
typedef __attribute__((ext_vector_type(8)))  float  v8f;
typedef __attribute__((ext_vector_type(4)))  unsigned int u32x4;
typedef __attribute__((ext_vector_type(8)))  int i32x8;
typedef __attribute__((ext_vector_type(4)))  int i32x4;

#define WAVES  8
#define TM     64     // Lp tile rows per iteration
#define NBATCH 64
#define LD     512    // drug length
#define LP     1024   // target length
#define DD     128    // feature dim (drug & protein)
#define KMAN   64     // MAN latent K

#if __has_builtin(__builtin_amdgcn_tensor_load_to_lds) && \
    __has_builtin(__builtin_amdgcn_s_wait_tensorcnt)
#define USE_TDM 1
#else
#define USE_TDM 0
#endif

// LDS layout: all row strides are multiples of 8 bf16 (16B) so every
// fragment access is an aligned ds_load_b128 / ds_store_b128.
struct SMem {
  __bf16 WbT[128][136];   // W_b^T  : [dd][dp]
  __bf16 Wk [64][136];    // W_x, then W_p : [man][d]
  union {
    float  f32[TM][DD];   // 32KB fp32 staging tile (TDM destination)
    __bf16 bt [TM][136];  // bf16 view after in-place conversion
  } Tst;                  // target tile
  __bf16 G  [TM][136];    // G = T @ W_b : [lp][dd]
  __bf16 Pt [KMAN][72];   // P^T : [man][lp_local]
  __bf16 Ct [LD][72];     // C^T tile : [ld][lp_local]
  __bf16 C  [TM][520];    // C tile   : [lp_local][ld]
  __bf16 Xt [KMAN][520];  // X^T = (A @ W_x^T)^T : [man][ld]
  float  sp[LP];          // H_p score accumulators
  float  sc[LD];          // H_c score accumulators
  float  red[256];        // block reduction scratch
};

__device__ __forceinline__ float fast_tanh(float x) {
#if __has_builtin(__builtin_amdgcn_tanhf)
  return __builtin_amdgcn_tanhf(x);
#elif __has_builtin(__builtin_amdgcn_tanh_f32)
  return __builtin_amdgcn_tanh_f32(x);
#else
  return tanhf(x);
#endif
}

__device__ __forceinline__ v16bf pack16(bf16x8 lo, bf16x8 hi) {
  v16bf r;
#pragma unroll
  for (int i = 0; i < 8; ++i) { r[i] = lo[i]; r[i + 8] = hi[i]; }
  return r;
}

// A-fragment (16x32 bf16): lane m = lane&15, half h = lane>>4.
// element e -> k = 16*(e>>3) + 8*h + (e&7)  => two contiguous 16B chunks.
__device__ __forceinline__ v16bf ldsA(const __bf16* rowbase, int k0, int h) {
  const __bf16* p = rowbase + k0 + 8 * h;
  return pack16(*(const bf16x8*)p, *(const bf16x8*)(p + 16));
}

// B-fragment (32x16 bf16) from B^T storage [N][K]: lane n = lane&15,
// element e -> k = 16*(e>>3) + 8*h + (e&7)  => two contiguous 16B chunks.
__device__ __forceinline__ v16bf ldsB(const __bf16* rowbase, int k0, int h) {
  const __bf16* p = rowbase + k0 + 8 * h;
  return pack16(*(const bf16x8*)p, *(const bf16x8*)(p + 16));
}

// A-fragment gathered directly from global fp32 (row-major [M][K]), cvt->bf16.
__device__ __forceinline__ v16bf glbA(const float* rowbase, int k0, int h) {
  const float4* p = (const float4*)(rowbase + k0 + 8 * h);
  const float4* q = (const float4*)(rowbase + k0 + 16 + 8 * h);
  float4 a0 = p[0], a1 = p[1], b0 = q[0], b1 = q[1];
  bf16x8 lo, hi;
  lo[0] = (__bf16)a0.x; lo[1] = (__bf16)a0.y; lo[2] = (__bf16)a0.z; lo[3] = (__bf16)a0.w;
  lo[4] = (__bf16)a1.x; lo[5] = (__bf16)a1.y; lo[6] = (__bf16)a1.z; lo[7] = (__bf16)a1.w;
  hi[0] = (__bf16)b0.x; hi[1] = (__bf16)b0.y; hi[2] = (__bf16)b0.z; hi[3] = (__bf16)b0.w;
  hi[4] = (__bf16)b1.x; hi[5] = (__bf16)b1.y; hi[6] = (__bf16)b1.z; hi[7] = (__bf16)b1.w;
  return pack16(lo, hi);
}

__device__ __forceinline__ v8f wmma_bf16(v16bf a, v16bf b, v8f c) {
  // D = A*B + C ; 16x16x32 bf16 -> f32
  return __builtin_amdgcn_wmma_f32_16x16x32_bf16(false, a, false, b, (short)0, c,
                                                 false, false);
}

// Generic LDS GEMM: A [M][lda] row-major (k contiguous), Bt [N][ldb] row-major
// (k contiguous), result stored TRANSPOSED: Dt[n][m] (contiguous b128 store).
template <int MTILES, int NTILES, int KSTEPS>
__device__ __forceinline__ void gemm_lds_tr(const __bf16* A, int lda,
                                            const __bf16* Bt, int ldb,
                                            __bf16* Dt, int ldd,
                                            int wave, int lane) {
  const int mn = lane & 15;
  const int h  = lane >> 4;
  for (int g = wave; g < MTILES * NTILES; g += WAVES) {
    const int mt = g / NTILES, nt = g % NTILES;
    v8f acc = {};
#pragma unroll
    for (int ks = 0; ks < KSTEPS; ++ks)
      acc = wmma_bf16(ldsA(A + (mt * 16 + mn) * lda, ks * 32, h),
                      ldsB(Bt + (nt * 16 + mn) * ldb, ks * 32, h), acc);
    bf16x8 o;
#pragma unroll
    for (int v = 0; v < 8; ++v) o[v] = (__bf16)acc[v];
    *(bf16x8*)(Dt + (nt * 16 + mn) * ldd + mt * 16 + 8 * h) = o;
  }
}

__device__ __forceinline__ void block_softmax(float* s, int n, float* red, int tid) {
  float mx = -1e30f;
  for (int i = tid; i < n; i += 256) mx = fmaxf(mx, s[i]);
  red[tid] = mx;
  __syncthreads();
  for (int off = 128; off > 0; off >>= 1) {
    if (tid < off) red[tid] = fmaxf(red[tid], red[tid + off]);
    __syncthreads();
  }
  mx = red[0];
  __syncthreads();
  float sum = 0.f;
  for (int i = tid; i < n; i += 256) { float e = __expf(s[i] - mx); s[i] = e; sum += e; }
  red[tid] = sum;
  __syncthreads();
  for (int off = 128; off > 0; off >>= 1) {
    if (tid < off) red[tid] += red[tid + off];
    __syncthreads();
  }
  const float inv = 1.f / red[0];
  __syncthreads();
  for (int i = tid; i < n; i += 256) s[i] *= inv;
  __syncthreads();
}

__global__ __launch_bounds__(256) void man_fused_kernel(
    const float* __restrict__ drug, const float* __restrict__ target,
    const float* __restrict__ W_b, const float* __restrict__ W_x,
    const float* __restrict__ W_p, const float* __restrict__ w_hx,
    const float* __restrict__ w_hp, float* __restrict__ out) {
  __shared__ SMem sm;
  const int b    = blockIdx.x;
  const int tid  = threadIdx.x;
  const int wave = tid >> 5;
  const int lane = tid & 31;
  const int mn   = lane & 15;
  const int h    = lane >> 4;
  const float* drug_b   = drug + (size_t)b * LD * DD;
  const float* target_b = target + (size_t)b * LP * DD;

  // ---------------- Phase 0: weights + X^T = (A @ W_x^T)^T ----------------
  for (int i = tid; i < 128 * 128; i += 256)
    sm.WbT[i & 127][i >> 7] = (__bf16)W_b[i];            // transpose W_b
  for (int i = tid; i < 64 * 128; i += 256)
    sm.Wk[i >> 7][i & 127] = (__bf16)W_x[i];             // W_x row-major
  for (int i = tid; i < LP; i += 256) sm.sp[i] = 0.f;
  for (int i = tid; i < LD; i += 256) sm.sc[i] = 0.f;
  __syncthreads();

  // X GEMM: M=512(ld), N=64(man), K=128(dd); A from global drug.
  for (int g = wave; g < 32 * 4; g += WAVES) {
    const int mt = g >> 2, nt = g & 3;
    v8f acc = {};
#pragma unroll
    for (int ks = 0; ks < 4; ++ks)
      acc = wmma_bf16(glbA(drug_b + (mt * 16 + mn) * DD, ks * 32, h),
                      ldsB(&sm.Wk[nt * 16 + mn][0], ks * 32, h), acc);
    bf16x8 o;
#pragma unroll
    for (int v = 0; v < 8; ++v) o[v] = (__bf16)acc[v];
    *(bf16x8*)(&sm.Xt[nt * 16 + mn][mt * 16 + 8 * h]) = o;
  }
  __syncthreads();

  // swap in W_p
  for (int i = tid; i < 64 * 128; i += 256)
    sm.Wk[i >> 7][i & 127] = (__bf16)W_p[i];

  v8f Uacc[16];
  {
    v8f z = {};
#pragma unroll
    for (int t = 0; t < 16; ++t) Uacc[t] = z;
  }
  __syncthreads();

  // ---------------- Phase 1: stream Lp in TM-row tiles ----------------
  for (int it = 0; it < LP / TM; ++it) {
    const int lp0 = it * TM;

    // ---- target tile -> LDS fp32 staging (Tensor Data Mover) ----
#if USE_TDM
    if (wave == 0) {
      const unsigned long long ga =
          (unsigned long long)(uintptr_t)(target_b + (size_t)lp0 * DD);
      u32x4 g0;
      g0[0] = 1u;                                          // count=1, load-descr
      g0[1] = (unsigned int)__builtin_offsetof(SMem, Tst); // lds_addr (bytes)
      g0[2] = (unsigned int)ga;                            // global_addr[31:0]
      g0[3] = (unsigned int)((ga >> 32) & 0x1FFFFFFu) | 0x80000000u; // addr hi | type=2
      i32x8 g1;
      g1[0] = 0x00020000;          // data_size = 4 bytes
      g1[1] = (int)(128u << 16);   // tensor_dim0 = 128 (bits 79:48 lo)
      g1[2] = (int)(1024u << 16);  // tensor_dim1 = 1024 (bits 111:80 lo)
      g1[3] = (int)(128u << 16);   // tile_dim0 = 128
      g1[4] = 64;                  // tile_dim1 = 64, tile_dim2 = 0
      g1[5] = 128;                 // tensor_dim0_stride = 128
      g1[6] = 0;                   // dim0_stride hi / dim1_stride lo
      g1[7] = 0;                   // dim1_stride hi
      i32x4 gz  = {0, 0, 0, 0};    // groups 2/3 unused (2D tensor)
      i32x8 gz8 = {0, 0, 0, 0, 0, 0, 0, 0};
      __builtin_amdgcn_tensor_load_to_lds(g0, g1, gz, gz, gz8, 0);
      __builtin_amdgcn_s_wait_tensorcnt(0);
    }
#else
    for (int i = tid; i < TM * DD; i += 256)
      sm.Tst.f32[i >> 7][i & 127] = target_b[(lp0 + (i >> 7)) * DD + (i & 127)];
#endif
    if (it + 1 < LP / TM)
      __builtin_prefetch((const void*)(target_b + (lp0 + TM) * DD + tid * 16), 0, 1);
    __syncthreads();

    // ---- in-place fp32 -> bf16 conversion of the tile ----
    {
      const int row = tid >> 2, col0 = (tid & 3) * 32;
      float4 f[8];
      const float4* src = (const float4*)&sm.Tst.f32[row][col0];
#pragma unroll
      for (int j = 0; j < 8; ++j) f[j] = src[j];
      __syncthreads();
#pragma unroll
      for (int j = 0; j < 4; ++j) {
        bf16x8 o;
        o[0] = (__bf16)f[2 * j].x;     o[1] = (__bf16)f[2 * j].y;
        o[2] = (__bf16)f[2 * j].z;     o[3] = (__bf16)f[2 * j].w;
        o[4] = (__bf16)f[2 * j + 1].x; o[5] = (__bf16)f[2 * j + 1].y;
        o[6] = (__bf16)f[2 * j + 1].z; o[7] = (__bf16)f[2 * j + 1].w;
        *(bf16x8*)&sm.Tst.bt[row][col0 + 8 * j] = o;
      }
    }
    __syncthreads();

    // G^T GEMM: M=128(dd), N=64(lp), K=128(dp) -> sm.G[lp][dd]
    gemm_lds_tr<8, 4, 4>(&sm.WbT[0][0], 136, &sm.Tst.bt[0][0], 136, &sm.G[0][0],
                         136, wave, lane);
    // P GEMM:  M=64(lp), N=64(man), K=128(dp) -> sm.Pt[man][lp]
    gemm_lds_tr<4, 4, 4>(&sm.Tst.bt[0][0], 136, &sm.Wk[0][0], 136, &sm.Pt[0][0],
                         72, wave, lane);
    __syncthreads();

    // C tile: tanh(A @ G^T): M=512(ld), N=64(lp), K=128(dd); A from global.
    for (int g = wave; g < 32 * 4; g += WAVES) {
      const int mt = g >> 2, nt = g & 3;
      v8f acc = {};
#pragma unroll
      for (int ks = 0; ks < 4; ++ks)
        acc = wmma_bf16(glbA(drug_b + (mt * 16 + mn) * DD, ks * 32, h),
                        ldsB(&sm.G[nt * 16 + mn][0], ks * 32, h), acc);
      bf16x8 o;
#pragma unroll
      for (int v = 0; v < 8; ++v) o[v] = (__bf16)fast_tanh(acc[v]);
      *(bf16x8*)(&sm.C[nt * 16 + mn][mt * 16 + 8 * h]) = o;  // C[lp][ld]
#pragma unroll
      for (int v = 0; v < 8; ++v)
        sm.Ct[mt * 16 + 8 * h + v][nt * 16 + mn] = o[v];     // Ct[ld][lp]
    }
    __syncthreads();

    // U += Ct @ P : M=512(ld), N=64(man), K=TM; persistent VGPR accumulators.
#pragma unroll
    for (int t = 0; t < 16; ++t) {
      const int g  = t * 8 + wave;
      const int mt = g >> 2, nt = g & 3;
#pragma unroll
      for (int ks = 0; ks < 2; ++ks)
        Uacc[t] = wmma_bf16(ldsA(&sm.Ct[mt * 16 + mn][0], ks * 32, h),
                            ldsB(&sm.Pt[nt * 16 + mn][0], ks * 32, h), Uacc[t]);
    }

    // V = C @ X : M=TM(lp), N=64(man), K=512(ld); fold into s_p immediately.
    for (int g = wave; g < 16; g += WAVES) {
      const int mt = g >> 2, nt = g & 3;
      v8f acc = {};
#pragma unroll
      for (int ks = 0; ks < 16; ++ks)
        acc = wmma_bf16(ldsA(&sm.C[mt * 16 + mn][0], ks * 32, h),
                        ldsB(&sm.Xt[nt * 16 + mn][0], ks * 32, h), acc);
      const int   man   = nt * 16 + mn;
      const float whp_n = w_hp[man];
      bf16x8 p8 = *(const bf16x8*)(&sm.Pt[man][mt * 16 + 8 * h]);
#pragma unroll
      for (int v = 0; v < 8; ++v) {
        const float val = fast_tanh(acc[v] + (float)p8[v]) * whp_n;
        atomicAdd(&sm.sp[lp0 + mt * 16 + 8 * h + v], val);
      }
    }
    __syncthreads();
  }

  // ---------------- Phase 2: H_c scores from U accumulators ----------------
#pragma unroll
  for (int t = 0; t < 16; ++t) {
    const int g  = t * 8 + wave;
    const int mt = g >> 2, nt = g & 3;
    const int   man   = nt * 16 + mn;
    const float whx_n = w_hx[man];
    bf16x8 x8 = *(const bf16x8*)(&sm.Xt[man][mt * 16 + 8 * h]);
#pragma unroll
    for (int v = 0; v < 8; ++v) {
      const float val = fast_tanh(Uacc[t][v] + (float)x8[v]) * whx_n;
      atomicAdd(&sm.sc[mt * 16 + 8 * h + v], val);
    }
  }
  __syncthreads();

  // ---------------- Phase 3: softmaxes + pooled outputs ----------------
  block_softmax(sm.sc, LD, sm.red, tid);
  block_softmax(sm.sp, LP, sm.red, tid);

  {
    const int dd = tid & 127, part = tid >> 7;
    float a = 0.f;
    for (int l = part * 256; l < part * 256 + 256; ++l)
      a += sm.sc[l] * drug_b[l * DD + dd];
    sm.red[tid] = a;
    __syncthreads();
    if (tid < 128) out[b * DD + tid] = sm.red[tid] + sm.red[tid + 128];
    __syncthreads();

    float p = 0.f;
    for (int l = part * 512; l < part * 512 + 512; ++l)
      p += sm.sp[l] * target_b[l * DD + dd];
    sm.red[tid] = p;
    __syncthreads();
    if (tid < 128)
      out[NBATCH * DD + b * DD + tid] = sm.red[tid] + sm.red[tid + 128];
  }
}

extern "C" void kernel_launch(void* const* d_in, const int* in_sizes, int n_in,
                              void* d_out, int out_size, void* d_ws, size_t ws_size,
                              hipStream_t stream) {
  (void)in_sizes; (void)n_in; (void)d_ws; (void)ws_size; (void)out_size;
  man_fused_kernel<<<dim3(NBATCH), dim3(256), 0, stream>>>(
      (const float*)d_in[0], (const float*)d_in[1], (const float*)d_in[2],
      (const float*)d_in[3], (const float*)d_in[4], (const float*)d_in[5],
      (const float*)d_in[6], (float*)d_out);
}
